// LNGRU_74655121539655
// MI455X (gfx1250) — matrix operand order repaired and checked by reading, not compiled, not run on hardware
//
#include <hip/hip_runtime.h>
#include <hip/hip_bf16.h>

// ---------------------------------------------------------------------------
// Bidirectional LayerNorm-GRU for MI455X (gfx1250, wave32, WMMA).
//   Phase 1: XW = LN(x @ W) for all t, both dirs  (parallel GEMM, wmma bf16)
//   Phase 2: persistent per-direction workgroup runs the 2048-step recurrence
//            h@U (wmma bf16) + LN + gates entirely through LDS.
// Latency tricks: double-buffered B fragments (no per-wmma loadcnt stall),
// global_prefetch of next step's xw block, non-temporal output stores.
// ---------------------------------------------------------------------------

typedef __bf16 v16bf __attribute__((ext_vector_type(16)));
typedef __bf16 v8bf  __attribute__((ext_vector_type(8)));
typedef float  v8f   __attribute__((ext_vector_type(8)));

#define T_STEPS 2048
#define BATCH   32
#define IN      256
#define HID     256
#define N3H     768
#define LDPAD   772   // f32 LDS row stride for [32][768] staging (bank spread)
#define HBF_LD  264   // bf16 LDS row stride for A operand (bank spread)

// --- ISA 16-bit A 16x32 fragment: lane L holds row M=L%16; two 8-half chunks
//     at K = (L<16?0:8) and K+16 (VGPRs 0-3 / 4-7). Row-major source.
static __device__ __forceinline__ v16bf ld_a_frag(const __bf16* tile, int ld, int lane) {
  const int r  = lane & 15;
  const int kc = (lane & 16) ? 8 : 0;
  const __bf16* p = tile + r * ld + kc;
  v8bf lo = *(const v8bf*)(p);
  v8bf hi = *(const v8bf*)(p + 16);
  return __builtin_shufflevector(lo, hi, 0,1,2,3,4,5,6,7,8,9,10,11,12,13,14,15);
}

// --- ISA 16-bit B 32x16 fragment: lane L holds col N=L%16; two 8-half chunks
//     at K = (L<16?0:8) and K+16, matching the A fragment. Source is B^T
//     row-major [N][K].
static __device__ __forceinline__ v16bf ld_b_frag(const __bf16* tileT, int ld, int lane) {
  const int n  = lane & 15;
  const int kc = (lane & 16) ? 8 : 0;
  const __bf16* p = tileT + n * ld + kc;
  v8bf lo = *(const v8bf*)(p);
  v8bf hi = *(const v8bf*)(p + 16);
  return __builtin_shufflevector(lo, hi, 0,1,2,3,4,5,6,7,8,9,10,11,12,13,14,15);
}

// --- f32 C/D 16x16 layout: VGPR v, lanes 0-15 -> (M=v, N=lane); 16-31 -> (M=8+v).
static __device__ __forceinline__ void st_c_tile(float* dst, int ld, int lane, v8f c) {
  const int n  = lane & 15;
  const int r0 = (lane & 16) ? 8 : 0;
#pragma unroll
  for (int v = 0; v < 8; ++v) dst[(size_t)(r0 + v) * ld + n] = c[v];
}

static __device__ __forceinline__ float sigmoidf_(float x) {
  return 1.0f / (1.0f + __expf(-x));
}

// ---------------------------------------------------------------------------
// Conversions
// ---------------------------------------------------------------------------
__global__ void k_cvt_x(const float* __restrict__ x, __bf16* __restrict__ xbf, int n) {
  int i = blockIdx.x * 256 + threadIdx.x;
  if (i < n) xbf[i] = (__bf16)x[i];
}

// Transpose+convert four [256,768] matrices -> bf16 [768,256]
__global__ void k_cvt_wu(const float* __restrict__ Wf, const float* __restrict__ Uf,
                         const float* __restrict__ Wb, const float* __restrict__ Ub,
                         __bf16* __restrict__ wt, __bf16* __restrict__ ut) {
  int i = blockIdx.x * 256 + threadIdx.x;            // 0 .. 4*256*768
  int which = i / (IN * N3H);
  int rem   = i - which * (IN * N3H);
  int k = rem / N3H;
  int n = rem - k * N3H;
  const float* src = (which == 0) ? Wf : (which == 1) ? Uf : (which == 2) ? Wb : Ub;
  __bf16* dst = (which == 0) ? wt : (which == 1) ? ut
              : (which == 2) ? (wt + (size_t)N3H * IN) : (ut + (size_t)N3H * IN);
  dst[(size_t)n * IN + k] = (__bf16)src[rem];
}

// ---------------------------------------------------------------------------
// Phase 1: XWln[d][t][b][0:768] = LN(x[t,b,:] @ W_d) * Wg + Wb   (bf16 out)
// grid: (T*B/32, 2), block 512 (16 waves). A-tile staged in LDS; each wave:
// 6 16x16 tiles, K-loop 8, B fragments double-buffered.
// ---------------------------------------------------------------------------
__global__ __launch_bounds__(512)
void k_xw_ln(const __bf16* __restrict__ xbf, const __bf16* __restrict__ wt,
             const float* __restrict__ Wg_f, const float* __restrict__ Wb_f,
             const float* __restrict__ Wg_b, const float* __restrict__ Wb_b,
             __bf16* __restrict__ xwln) {
  __shared__ __align__(16) float  s_acc[BATCH * LDPAD];    // 98,816 B
  __shared__ __align__(16) __bf16 s_a[BATCH * HBF_LD];     // 16,896 B
  __shared__ float s_part[BATCH * 16 * 2];
  __shared__ float s_stat[BATCH * 2];

  const int d    = blockIdx.y;
  const int m0   = blockIdx.x * 32;
  const int tid  = threadIdx.x;
  const int lane = tid & 31;
  const int w    = tid >> 5;
  const int rt   = w & 1;    // row tile 0/1
  const int cg   = w >> 1;   // column group 0..7 (6 tiles each)
  const __bf16* wtd = wt + (size_t)d * N3H * IN;
  const float* gam = d ? Wg_b : Wg_f;
  const float* bet = d ? Wb_b : Wb_f;

  // stage A tile (32x256 bf16) in LDS: thread copies 16 contiguous elements
  {
    int idx = tid * 16;
    int row = idx >> 8, col = idx & 255;
    const __bf16* src = xbf + (size_t)(m0 + row) * IN + col;
    v8bf a0 = *(const v8bf*)(src);
    v8bf a1 = *(const v8bf*)(src + 8);
    *(v8bf*)(&s_a[row * HBF_LD + col])     = a0;
    *(v8bf*)(&s_a[row * HBF_LD + col + 8]) = a1;
  }
  __syncthreads();

  const v8f vz = {0.f, 0.f, 0.f, 0.f, 0.f, 0.f, 0.f, 0.f};
  v8f c[6];
#pragma unroll
  for (int i = 0; i < 6; ++i) c[i] = vz;

  v16bf bufA[6], bufB[6];
#pragma unroll
  for (int i = 0; i < 6; ++i)
    bufA[i] = ld_b_frag(wtd + (size_t)((cg * 6 + i) * 16) * IN, IN, lane);

#pragma unroll
  for (int kb = 0; kb < 8; ++kb) {
    v16bf a = ld_a_frag(&s_a[(size_t)(rt * 16) * HBF_LD + kb * 32], HBF_LD, lane);
    if (kb < 7) {
#pragma unroll
      for (int i = 0; i < 6; ++i) {
        v16bf nb = ld_b_frag(wtd + (size_t)((cg * 6 + i) * 16) * IN + (kb + 1) * 32,
                             IN, lane);
        if (kb & 1) bufA[i] = nb; else bufB[i] = nb;
      }
    }
#pragma unroll
    for (int i = 0; i < 6; ++i) {
      v16bf b = (kb & 1) ? bufB[i] : bufA[i];
      c[i] = __builtin_amdgcn_wmma_f32_16x16x32_bf16(false, a, false, b,
                                                     (short)0, c[i], false, false);
    }
  }
#pragma unroll
  for (int i = 0; i < 6; ++i)
    st_c_tile(&s_acc[(size_t)(rt * 16) * LDPAD + (cg * 6 + i) * 16], LDPAD, lane, c[i]);
  __syncthreads();

  // row stats over 768 (16 threads/row)
  {
    int row = tid >> 4, seg = tid & 15;
    float s = 0.f, sq = 0.f;
    for (int jj = 0; jj < 48; ++jj) {
      float v = s_acc[row * LDPAD + jj * 16 + seg];
      s += v; sq += v * v;
    }
    s_part[(row * 16 + seg) * 2 + 0] = s;
    s_part[(row * 16 + seg) * 2 + 1] = sq;
  }
  __syncthreads();
  if (tid < 32) {
    float s = 0.f, sq = 0.f;
    for (int k = 0; k < 16; ++k) {
      s  += s_part[(tid * 16 + k) * 2 + 0];
      sq += s_part[(tid * 16 + k) * 2 + 1];
    }
    float mean = s * (1.f / 768.f);
    float var  = sq * (1.f / 768.f) - mean * mean;
    s_stat[tid * 2 + 0] = mean;
    s_stat[tid * 2 + 1] = rsqrtf(var + 1e-5f);
  }
  __syncthreads();

  {
    int row = tid >> 4, seg = tid & 15;
    float mean = s_stat[row * 2], rstd = s_stat[row * 2 + 1];
    __bf16* dst = xwln + ((size_t)d * T_STEPS * BATCH + m0 + row) * N3H;
    for (int jj = 0; jj < 48; ++jj) {
      int j = jj * 16 + seg;
      float v = (s_acc[row * LDPAD + j] - mean) * rstd * gam[j] + bet[j];
      dst[j] = (__bf16)v;
    }
  }
}

// ---------------------------------------------------------------------------
// Phase 2: persistent recurrence. grid=2 (dir), block 512. U^T streamed from
// L2 each step (double-buffered into regs); h kept in LDS (f32 + bf16 operand).
// ---------------------------------------------------------------------------
__global__ __launch_bounds__(512)
void k_recurrent(const __bf16* __restrict__ ut, const __bf16* __restrict__ xwln,
                 const float* __restrict__ h0,
                 const float* __restrict__ b_f, const float* __restrict__ Ug_f,
                 const float* __restrict__ Ub_f,
                 const float* __restrict__ b_b, const float* __restrict__ Ug_b,
                 const float* __restrict__ Ub_b,
                 float* __restrict__ out) {
  __shared__ __align__(16) float  s_hu[BATCH * LDPAD];     // 98,816 B
  __shared__ __align__(16) __bf16 s_hbf[BATCH * HBF_LD];   // 16,896 B
  __shared__ __align__(16) float  s_hf[BATCH * HID];       // 32,768 B
  __shared__ float s_part[BATCH * 16 * 2];                 //  4,096 B
  __shared__ float s_stat[BATCH * 2];

  const int d    = blockIdx.x;
  const int tid  = threadIdx.x;
  const int lane = tid & 31;
  const int w    = tid >> 5;
  const int rt   = w & 1;
  const int cg   = w >> 1;
  const __bf16* utd = ut + (size_t)d * N3H * IN;
  const __bf16* xwd = xwln + (size_t)d * T_STEPS * BATCH * N3H;
  const float* bias = d ? b_b  : b_f;
  const float* gam  = d ? Ug_b : Ug_f;
  const float* bet  = d ? Ub_b : Ub_f;

  for (int e = tid; e < BATCH * HID; e += 512) {
    int row = e >> 8, j = e & 255;
    float v = h0[(size_t)d * BATCH * HID + e];
    s_hf[row * HID + j]     = v;
    s_hbf[row * HBF_LD + j] = (__bf16)v;
  }
  // warm the first step's xw block into the cache hierarchy
  {
    const int te0 = d ? (T_STEPS - 1) : 0;
    __builtin_prefetch((const char*)(xwd + (size_t)te0 * BATCH * N3H) + tid * 96, 0, 3);
  }
  __syncthreads();

  const v8f vz = {0.f, 0.f, 0.f, 0.f, 0.f, 0.f, 0.f, 0.f};

  for (int t = 0; t < T_STEPS; ++t) {
    const int te = d ? (T_STEPS - 1 - t) : t;

    // prefetch next step's xw block (48 KB) while this step's GEMM runs
    if (t + 1 < T_STEPS) {
      const int ten = d ? (T_STEPS - 2 - t) : (t + 1);
      __builtin_prefetch((const char*)(xwd + (size_t)ten * BATCH * N3H) + tid * 96, 0, 3);
    }

    // ---- hu = h @ U : 48 wmma per wave, B double-buffered ----
    v8f c[6];
#pragma unroll
    for (int i = 0; i < 6; ++i) c[i] = vz;

    v16bf bufA[6], bufB[6];
#pragma unroll
    for (int i = 0; i < 6; ++i)
      bufA[i] = ld_b_frag(utd + (size_t)((cg * 6 + i) * 16) * IN, IN, lane);

#pragma unroll
    for (int kb = 0; kb < 8; ++kb) {
      v16bf a = ld_a_frag(&s_hbf[(size_t)(rt * 16) * HBF_LD + kb * 32], HBF_LD, lane);
      if (kb < 7) {
#pragma unroll
        for (int i = 0; i < 6; ++i) {
          v16bf nb = ld_b_frag(utd + (size_t)((cg * 6 + i) * 16) * IN + (kb + 1) * 32,
                               IN, lane);
          if (kb & 1) bufA[i] = nb; else bufB[i] = nb;
        }
      }
#pragma unroll
      for (int i = 0; i < 6; ++i) {
        v16bf b = (kb & 1) ? bufB[i] : bufA[i];
        c[i] = __builtin_amdgcn_wmma_f32_16x16x32_bf16(false, a, false, b,
                                                       (short)0, c[i], false, false);
      }
    }
#pragma unroll
    for (int i = 0; i < 6; ++i)
      st_c_tile(&s_hu[(size_t)(rt * 16) * LDPAD + (cg * 6 + i) * 16], LDPAD, lane, c[i]);
    __syncthreads();

    // ---- LN stats over 768 per row ----
    {
      int row = tid >> 4, seg = tid & 15;
      float s = 0.f, sq = 0.f;
      for (int jj = 0; jj < 48; ++jj) {
        float v = s_hu[row * LDPAD + jj * 16 + seg];
        s += v; sq += v * v;
      }
      s_part[(row * 16 + seg) * 2 + 0] = s;
      s_part[(row * 16 + seg) * 2 + 1] = sq;
    }
    __syncthreads();
    if (tid < 32) {
      float s = 0.f, sq = 0.f;
      for (int k = 0; k < 16; ++k) {
        s  += s_part[(tid * 16 + k) * 2 + 0];
        sq += s_part[(tid * 16 + k) * 2 + 1];
      }
      float mean = s * (1.f / 768.f);
      float var  = sq * (1.f / 768.f) - mean * mean;
      s_stat[tid * 2 + 0] = mean;
      s_stat[tid * 2 + 1] = rsqrtf(var + 1e-5f);
    }
    __syncthreads();

    // ---- gates + state update + output store ----
    {
      int row = tid >> 4, seg = tid & 15;
      float mean = s_stat[row * 2], rstd = s_stat[row * 2 + 1];
      const __bf16* xw = xwd + ((size_t)te * BATCH + row) * N3H;
      float* orow = out + ((size_t)te * BATCH + row) * (2 * HID) + d * HID;
#pragma unroll 4
      for (int jj = 0; jj < 16; ++jj) {
        int j = jj * 16 + seg;  // consecutive lanes -> consecutive j
        float hu0 = (s_hu[row * LDPAD + j]       - mean) * rstd * gam[j]       + bet[j];
        float hu1 = (s_hu[row * LDPAD + 256 + j] - mean) * rstd * gam[256 + j] + bet[256 + j];
        float hu2 = (s_hu[row * LDPAD + 512 + j] - mean) * rstd * gam[512 + j] + bet[512 + j];
        float xw0 = (float)xw[j];
        float xw1 = (float)xw[256 + j];
        float xw2 = (float)xw[512 + j];
        float z  = sigmoidf_(xw0 + hu0 + bias[j]);
        float r  = sigmoidf_(xw1 + hu1 + bias[256 + j]);
        float hc = tanhf(r * hu2 + xw2 + bias[512 + j]);
        float hn = (1.f - z) * hc + z * s_hf[row * HID + j];
        s_hf[row * HID + j]     = hn;
        s_hbf[row * HBF_LD + j] = (__bf16)hn;
        __builtin_nontemporal_store(hn, &orow[j]);   // streamed once; keep U hot in L2
      }
    }
    __syncthreads();
  }

  // final hidden state -> out[T*B*2H + d*B*H ...]
  for (int e = tid; e < BATCH * HID; e += 512)
    out[(size_t)T_STEPS * BATCH * 2 * HID + (size_t)d * BATCH * HID + e] = s_hf[e];
}

// ---------------------------------------------------------------------------
extern "C" void kernel_launch(void* const* d_in, const int* in_sizes, int n_in,
                              void* d_out, int out_size, void* d_ws, size_t ws_size,
                              hipStream_t stream) {
  (void)in_sizes; (void)n_in; (void)out_size; (void)ws_size;
  const float* x    = (const float*)d_in[0];
  const float* h0   = (const float*)d_in[1];
  const float* W_f  = (const float*)d_in[2];
  const float* U_f  = (const float*)d_in[3];
  const float* b_f  = (const float*)d_in[4];
  const float* Wg_f = (const float*)d_in[5];
  const float* Wb_f = (const float*)d_in[6];
  const float* Ug_f = (const float*)d_in[7];
  const float* Ub_f = (const float*)d_in[8];
  const float* W_b  = (const float*)d_in[9];
  const float* U_b  = (const float*)d_in[10];
  const float* b_b  = (const float*)d_in[11];
  const float* Wg_b = (const float*)d_in[12];
  const float* Wb_b = (const float*)d_in[13];
  const float* Ug_b = (const float*)d_in[14];
  const float* Ub_b = (const float*)d_in[15];
  float* out = (float*)d_out;

  // workspace layout (bytes): xbf 33,554,432 | Wt 786,432 | Ut 786,432 | xwln 201,326,592
  char* ws = (char*)d_ws;
  __bf16* xbf  = (__bf16*)(ws);
  __bf16* wt   = (__bf16*)(ws + 33554432);
  __bf16* ut   = (__bf16*)(ws + 33554432 + 786432);
  __bf16* xwln = (__bf16*)(ws + 33554432 + 2 * 786432);

  k_cvt_x<<<(T_STEPS * BATCH * IN) / 256, 256, 0, stream>>>(x, xbf, T_STEPS * BATCH * IN);
  k_cvt_wu<<<(4 * IN * N3H) / 256, 256, 0, stream>>>(W_f, U_f, W_b, U_b, wt, ut);
  dim3 g1(T_STEPS * BATCH / 32, 2);
  k_xw_ln<<<g1, 512, 0, stream>>>(xbf, wt, Wg_f, Wb_f, Wg_b, Wb_b, xwln);
  k_recurrent<<<2, 512, 0, stream>>>(ut, xwln, h0, b_f, Ug_f, Ub_f,
                                     b_b, Ug_b, Ub_b, out);
}
